// LorentzNet_89833535963778
// MI455X (gfx1250) — matrix operand order repaired and checked by reading, not compiled, not run on hardware
//
#include <hip/hip_runtime.h>

#define HID    72
#define NPART  128
#define NB     32
#define NNODE  (NB*NPART)     /* 4096 */
#define EPN    (NPART-1)      /* 127 contiguous edges per destination node */
#define NLAYER 6

typedef _Float16 h8  __attribute__((ext_vector_type(8)));
typedef _Float16 h16 __attribute__((ext_vector_type(16)));
typedef float    f8  __attribute__((ext_vector_type(8)));

static __device__ __forceinline__ h16 cat16(h8 lo, h8 hi) {
  return __builtin_shufflevector(lo, hi, 0,1,2,3,4,5,6,7,8,9,10,11,12,13,14,15);
}

static __device__ __forceinline__ f8 wmma16(h16 a, h16 b, f8 c) {
  // v_wmma_f32_16x16x32_f16 : D = A(16x32 f16) * B(32x16 f16) + C(16x16 f32)
  return __builtin_amdgcn_wmma_f32_16x16x32_f16(false, a, false, b, (short)0, c, false, false);
}

static __device__ __forceinline__ float psi_f(float v) {
  return copysignf(log1pf(fabsf(v)), v);
}

// ---------------------------------------------------------------------------
// Pack an f32 row-major weight W[Krows x Ncols] into f16 WMMA-B fragments.
// Fragment (kt,nt) holds B-tile rows K=kt*32..+31, cols N=nt*16..+15.
// B layout (16-bit, 32x16): lane L -> col n = nt*16 + (L&15), half = L>>4,
// element i (0..15) -> K = kt*32 + (i<8 ? half*8+i : 16+half*8+(i-8)).  Stored lane-contiguous so a
// lane's 16 halves are one 32-byte load.
// ---------------------------------------------------------------------------
__global__ void pack_wmma_b(const float* __restrict__ W, _Float16* __restrict__ out,
                            int Krows, int Ncols, int KT, int NT) {
  int id = blockIdx.x * blockDim.x + threadIdx.x;
  int total = KT * NT * 512;
  if (id >= total) return;
  int frag   = id >> 9;
  int within = id & 511;
  int lane   = within >> 4;
  int i      = within & 15;
  int kt = frag / NT, nt = frag - kt * NT;
  int ncol = nt * 16 + (lane & 15);
  int half = lane >> 4;
  int K = kt * 32 + (i < 8 ? half * 8 + i : 16 + half * 8 + (i - 8));
  float v = (K < Krows && ncol < Ncols) ? W[K * Ncols + ncol] : 0.0f;
  out[id] = (_Float16)v;
}

// ---------------------------------------------------------------------------
// h = scalars @ emb_W + emb_b (f32 master + f16 mirror) ;  x = mom4
// ---------------------------------------------------------------------------
__global__ void embed_kernel(const float* __restrict__ scalars,
                             const float* __restrict__ mom4,
                             const float* __restrict__ embW,
                             const float* __restrict__ embb,
                             float* __restrict__ hbuf, _Float16* __restrict__ h16buf,
                             float* __restrict__ xbuf) {
  int idx = blockIdx.x * blockDim.x + threadIdx.x;
  if (idx < NNODE * HID) {
    int r = idx / HID, f = idx - r * HID;
    float s = embb[f];
    #pragma unroll
    for (int k = 0; k < 4; ++k) s += scalars[r * 4 + k] * embW[k * HID + f];
    hbuf[idx]   = s;
    h16buf[idx] = (_Float16)s;
  }
  if (idx < NNODE * 4) xbuf[idx] = mom4[idx];
}

// ---------------------------------------------------------------------------
// Per-node edge kernel: block = node n, its 127 contiguous edges.
// 4 waves x 2 tiles of 16 edges. Per tile:
//   feat(16x160 f16) -> GEMM1(We1,relu) -> GEMM2(We2,relu)=m
//   w = sigmoid(m@Wm+bm);  wm[n] += w*m  (deterministic LDS reduce)
//   if compute_x: phix = relu(m@Wx1+bx1)@Wx2 ; num[n] += phix * x[ej]
// h is staged from the persistent f16 mirror: no per-edge f32->f16 converts.
// ---------------------------------------------------------------------------
__global__ __launch_bounds__(128) void lgeb_edge_kernel(
    const _Float16* __restrict__ h16buf, const float* __restrict__ x,
    const int* __restrict__ edge_j,
    const _Float16* __restrict__ We1p, const _Float16* __restrict__ We2p,
    const _Float16* __restrict__ Wx1p,
    const float* __restrict__ be1, const float* __restrict__ be2,
    const float* __restrict__ Wm,  const float* __restrict__ bm,
    const float* __restrict__ bx1, const float* __restrict__ Wx2,
    float* __restrict__ wm_out, float* __restrict__ num_out, int compute_x) {
  __shared__ _Float16 sFeat[4][16 * 160];   // feature tile; reused as m (16x96)
  __shared__ _Float16 sTmp [4][16 * 96];    // GEMM1 output
  __shared__ float    sW   [4][16];         // edge gates
  __shared__ float    sPhix[4][16];         // per-edge phix
  __shared__ float    sWmP [4][HID];        // per-wave wm partials
  __shared__ float    sNumP[4][4];          // per-wave num partials

  const int n    = blockIdx.x;
  const int tid  = threadIdx.x;
  const int wave = tid >> 5, lane = tid & 31;
  const int m16  = lane & 15, half = lane >> 4;
  const int gbase0 = n * EPN;

  float wmAcc[3] = {0.f, 0.f, 0.f};
  float na0 = 0.f, na1 = 0.f, na2 = 0.f, na3 = 0.f;

  // zero K-padding cols 80..95 of sTmp (GEMM1 only writes cols 0..79)
  for (int idx = lane; idx < 16 * 16; idx += 32)
    sTmp[wave][(idx >> 4) * 96 + 80 + (idx & 15)] = (_Float16)0.0f;
  __syncthreads();

  for (int tt = 0; tt < 2; ++tt) {
    const int tile  = tt * 4 + wave;
    const int ebase = tile * 16;              // local edge index base

    // ---- stage feature tile: [h_i(72) | h_j(72) | psi(norm) | psi(prod) | 0-pad]
    if (lane < 16) {
      int t = ebase + lane;
      float nrm = 0.f, prd = 0.f;
      if (t < EPN) {
        int gj = edge_j[gbase0 + t];
        const float4 xi = *(const float4*)(x + (size_t)n  * 4);
        const float4 xj = *(const float4*)(x + (size_t)gj * 4);
        float d0 = xi.x - xj.x, d1 = xi.y - xj.y, d2 = xi.z - xj.z, d3 = xi.w - xj.w;
        nrm = psi_f(d0 * d0 - d1 * d1 - d2 * d2 - d3 * d3);
        prd = psi_f(xi.x * xj.x - xi.y * xj.y - xi.z * xj.z - xi.w * xj.w);
      }
      _Float16* row = &sFeat[wave][lane * 160];
      row[144] = (_Float16)nrm;
      row[145] = (_Float16)prd;
      #pragma unroll
      for (int c = 146; c < 160; ++c) row[c] = (_Float16)0.0f;
    }
    // h_i / h_j rows as 16-byte f16 vector copies (9 chunks of 8 halves per row)
    for (int idx = lane; idx < 16 * 9; idx += 32) {
      int e = idx / 9, c = (idx - e * 9) * 8;
      int t = ebase + e;
      h8 vi = {}, vj = {};
      if (t < EPN) {
        int gj = edge_j[gbase0 + t];
        vi = *(const h8*)(h16buf + (size_t)n  * HID + c);
        vj = *(const h8*)(h16buf + (size_t)gj * HID + c);
      }
      *(h8*)&sFeat[wave][e * 160 + c]       = vi;
      *(h8*)&sFeat[wave][e * 160 + HID + c] = vj;
    }
    __syncthreads();

    // ---- GEMM1: feat(16x160) @ We1p(160x80) -> relu -> sTmp
    for (int nt = 0; nt < 5; ++nt) {
      f8 acc = {};
      for (int kt = 0; kt < 5; ++kt) {
        const _Float16* p = &sFeat[wave][m16 * 160 + kt * 32 + half * 8];
        h16 a = cat16(*(const h8*)p, *(const h8*)(p + 16));
        h16 b = *(const h16*)(We1p + ((size_t)(kt * 5 + nt) * 32 + lane) * 16);
        acc = wmma16(a, b, acc);
      }
      int   nc   = nt * 16 + m16;
      float bias = (nc < HID) ? be1[nc] : 0.0f;
      #pragma unroll
      for (int r = 0; r < 8; ++r) {
        int   row = r + 8 * half;
        float v   = acc[r] + bias;
        sTmp[wave][row * 96 + nc] = (_Float16)(v > 0.f ? v : 0.f);
      }
    }
    __syncthreads();

    // ---- GEMM2: sTmp(16x96) @ We2p(96x80) -> relu -> m (overwrites sFeat, 16x96)
    _Float16* sM = sFeat[wave];
    for (int idx = lane; idx < 16 * 16; idx += 32)          // zero K-pad cols 80..95
      sM[(idx >> 4) * 96 + 80 + (idx & 15)] = (_Float16)0.0f;
    for (int nt = 0; nt < 5; ++nt) {
      f8 acc = {};
      for (int kt = 0; kt < 3; ++kt) {
        const _Float16* p = &sTmp[wave][m16 * 96 + kt * 32 + half * 8];
        h16 a = cat16(*(const h8*)p, *(const h8*)(p + 16));
        h16 b = *(const h16*)(We2p + ((size_t)(kt * 5 + nt) * 32 + lane) * 16);
        acc = wmma16(a, b, acc);
      }
      int   nc   = nt * 16 + m16;
      float bias = (nc < HID) ? be2[nc] : 0.0f;
      #pragma unroll
      for (int r = 0; r < 8; ++r) {
        int   row = r + 8 * half;
        float v   = acc[r] + bias;
        sM[row * 96 + nc] = (_Float16)(v > 0.f ? v : 0.f);
      }
    }
    __syncthreads();

    // ---- edge gates: w = sigmoid(m @ Wm + bm)
    if (lane < 16) {
      int   t = ebase + lane;
      float s = bm[0];
      for (int f = 0; f < HID; ++f) s += (float)sM[lane * 96 + f] * Wm[f];
      sW[wave][lane] = (t < EPN) ? (1.0f / (1.0f + expf(-s))) : 0.0f;
    }
    __syncthreads();

    // ---- wm partial: sum_e w_e * m[e][f]  (deterministic sequential reduce)
    #pragma unroll
    for (int k = 0; k < 3; ++k) {
      int f = lane + 32 * k;
      if (f < HID) {
        float s = 0.f;
        for (int e = 0; e < 16; ++e) s += sW[wave][e] * (float)sM[e * 96 + f];
        wmAcc[k] += s;
      }
    }

    // ---- x path: phix = relu(m@Wx1+bx1) @ Wx2 ; num += phix * x[ej]
    if (compute_x) {
      float phiAcc[8] = {0.f,0.f,0.f,0.f,0.f,0.f,0.f,0.f};
      for (int nt = 0; nt < 5; ++nt) {
        f8 acc = {};
        for (int kt = 0; kt < 3; ++kt) {
          const _Float16* p = &sM[m16 * 96 + kt * 32 + half * 8];
          h16 a = cat16(*(const h8*)p, *(const h8*)(p + 16));
          h16 b = *(const h16*)(Wx1p + ((size_t)(kt * 5 + nt) * 32 + lane) * 16);
          acc = wmma16(a, b, acc);
        }
        int   nc   = nt * 16 + m16;
        float bias = (nc < HID) ? bx1[nc] : 0.0f;
        float wx2v = (nc < HID) ? Wx2[nc] : 0.0f;
        #pragma unroll
        for (int r = 0; r < 8; ++r) {
          float v = acc[r] + bias;
          v = (v > 0.f ? v : 0.f) * wx2v;
          #pragma unroll
          for (int d = 1; d < 16; d <<= 1) v += __shfl_xor(v, d, 16); // sum over 16 cols
          phiAcc[r] += v;
        }
      }
      if (m16 == 0) {
        #pragma unroll
        for (int r = 0; r < 8; ++r) sPhix[wave][r + 8 * half] = phiAcc[r];
      }
      __syncthreads();
      float c0 = 0.f, c1 = 0.f, c2 = 0.f, c3 = 0.f;
      if (lane < 16) {
        int t = ebase + lane;
        if (t < EPN) {
          int gj = edge_j[gbase0 + t];
          float p = sPhix[wave][lane];
          const float4 xj = *(const float4*)(x + (size_t)gj * 4);
          c0 = p * xj.x; c1 = p * xj.y; c2 = p * xj.z; c3 = p * xj.w;
        }
      }
      #pragma unroll
      for (int d = 1; d < 32; d <<= 1) {
        c0 += __shfl_xor(c0, d); c1 += __shfl_xor(c1, d);
        c2 += __shfl_xor(c2, d); c3 += __shfl_xor(c3, d);
      }
      if (lane == 0) { na0 += c0; na1 += c1; na2 += c2; na3 += c3; }
    }
    __syncthreads();
  }

  // ---- block reduction & output
  #pragma unroll
  for (int k = 0; k < 3; ++k) {
    int f = lane + 32 * k;
    if (f < HID) sWmP[wave][f] = wmAcc[k];
  }
  if (compute_x && lane == 0) {
    sNumP[wave][0] = na0; sNumP[wave][1] = na1;
    sNumP[wave][2] = na2; sNumP[wave][3] = na3;
  }
  __syncthreads();
  if (tid < HID)
    wm_out[(size_t)n * HID + tid] =
        sWmP[0][tid] + sWmP[1][tid] + sWmP[2][tid] + sWmP[3][tid];
  if (compute_x && tid < 4)
    num_out[(size_t)n * 4 + tid] =
        sNumP[0][tid] + sNumP[1][tid] + sNumP[2][tid] + sNumP[3][tid];
}

// ---------------------------------------------------------------------------
// Node update: h += relu([h,wm]@Wh1+bh1)@Wh2+bh2 ; x += C_W*num/127
// 64 blocks x 128 threads; each wave owns one 16-row tile.
// Also refreshes the persistent f16 mirror of h.
// ---------------------------------------------------------------------------
__global__ __launch_bounds__(128) void lgeb_node_kernel(
    float* __restrict__ hbuf, _Float16* __restrict__ h16buf,
    float* __restrict__ xbuf,
    const float* __restrict__ wm, const float* __restrict__ num,
    const _Float16* __restrict__ Wh1p, const _Float16* __restrict__ Wh2p,
    const float* __restrict__ bh1, const float* __restrict__ bh2, int compute_x) {
  __shared__ _Float16 sA[4][16 * 160];
  __shared__ _Float16 sT[4][16 * 96];
  const int tid  = threadIdx.x;
  const int wave = tid >> 5, lane = tid & 31;
  const int m16  = lane & 15, half = lane >> 4;
  const int r0   = (blockIdx.x * 4 + wave) * 16;

  for (int idx = lane; idx < 16 * 16; idx += 32)
    sT[wave][(idx >> 4) * 96 + 80 + (idx & 15)] = (_Float16)0.0f;
  for (int idx = lane; idx < 16 * 160; idx += 32) {
    int rl = idx / 160, c = idx - rl * 160;
    float v = 0.f;
    if (c < HID)          v = hbuf[(size_t)(r0 + rl) * HID + c];
    else if (c < 2 * HID) v = wm  [(size_t)(r0 + rl) * HID + (c - HID)];
    sA[wave][idx] = (_Float16)v;
  }
  __syncthreads();

  for (int nt = 0; nt < 5; ++nt) {
    f8 acc = {};
    for (int kt = 0; kt < 5; ++kt) {
      const _Float16* p = &sA[wave][m16 * 160 + kt * 32 + half * 8];
      h16 a = cat16(*(const h8*)p, *(const h8*)(p + 16));
      h16 b = *(const h16*)(Wh1p + ((size_t)(kt * 5 + nt) * 32 + lane) * 16);
      acc = wmma16(a, b, acc);
    }
    int   nc   = nt * 16 + m16;
    float bias = (nc < HID) ? bh1[nc] : 0.0f;
    #pragma unroll
    for (int r = 0; r < 8; ++r) {
      int   row = r + 8 * half;
      float v   = acc[r] + bias;
      sT[wave][row * 96 + nc] = (_Float16)(v > 0.f ? v : 0.f);
    }
  }
  __syncthreads();

  for (int nt = 0; nt < 5; ++nt) {
    f8 acc = {};
    for (int kt = 0; kt < 3; ++kt) {
      const _Float16* p = &sT[wave][m16 * 96 + kt * 32 + half * 8];
      h16 a = cat16(*(const h8*)p, *(const h8*)(p + 16));
      h16 b = *(const h16*)(Wh2p + ((size_t)(kt * 5 + nt) * 32 + lane) * 16);
      acc = wmma16(a, b, acc);
    }
    int nc = nt * 16 + m16;
    if (nc < HID) {
      float bias = bh2[nc];
      #pragma unroll
      for (int r = 0; r < 8; ++r) {
        int    row = r + 8 * half;
        size_t o   = (size_t)(r0 + row) * HID + nc;
        float  nv  = hbuf[o] + acc[r] + bias;
        hbuf[o]   = nv;
        h16buf[o] = (_Float16)nv;
      }
    }
  }
  if (compute_x) {
    const float cscale = 0.005f / 127.0f;   // C_W * mean over 127 edges
    for (int idx = lane; idx < 16 * 4; idx += 32) {
      int rl = idx >> 2, c = idx & 3;
      xbuf[(size_t)(r0 + rl) * 4 + c] += cscale * num[(size_t)(r0 + rl) * 4 + c];
    }
  }
}

// ---------------------------------------------------------------------------
// Decoder: z = mean_p(h*mask) ; out = relu(z@W1+b1)@W2+b2   (tiny)
// ---------------------------------------------------------------------------
__global__ __launch_bounds__(256) void decoder_kernel(
    const float* __restrict__ hbuf, const float* __restrict__ mask,
    const float* __restrict__ W1, const float* __restrict__ b1,
    const float* __restrict__ W2, const float* __restrict__ b2,
    float* __restrict__ out) {
  __shared__ float sZ[NB * HID];
  __shared__ float sH[NB * HID];
  int tid = threadIdx.x;
  for (int idx = tid; idx < NB * HID; idx += 256) {
    int b = idx / HID, f = idx - b * HID;
    float s = 0.f;
    for (int p = 0; p < NPART; ++p) {
      float mk = (mask[b * NPART + p] != 0.f) ? 1.f : 0.f;
      s += mk * hbuf[(size_t)(b * NPART + p) * HID + f];
    }
    sZ[idx] = s * (1.0f / (float)NPART);
  }
  __syncthreads();
  for (int idx = tid; idx < NB * HID; idx += 256) {
    int b = idx / HID, f = idx - b * HID;
    float s = b1[f];
    for (int k = 0; k < HID; ++k) s += sZ[b * HID + k] * W1[k * HID + f];
    sH[idx] = s > 0.f ? s : 0.f;
  }
  __syncthreads();
  if (tid < NB * 2) {
    int b = tid >> 1, o = tid & 1;
    float s = b2[o];
    for (int k = 0; k < HID; ++k) s += sH[b * HID + k] * W2[k * 2 + o];
    out[b * 2 + o] = s;
  }
}

// ---------------------------------------------------------------------------
extern "C" void kernel_launch(void* const* d_in, const int* in_sizes, int n_in,
                              void* d_out, int out_size, void* d_ws, size_t ws_size,
                              hipStream_t stream) {
  (void)in_sizes; (void)n_in; (void)out_size; (void)ws_size;
  const float* mom4    = (const float*)d_in[0];
  const float* mask    = (const float*)d_in[1];
  const float* scalars = (const float*)d_in[2];
  const int*   edge_i  = (const int*)d_in[3]; (void)edge_i; // structure: ei[n*127..] == n
  const int*   edge_j  = (const int*)d_in[4];
  const float* embW    = (const float*)d_in[5];
  const float* embb    = (const float*)d_in[6];
  const float *We1[NLAYER], *be1[NLAYER], *We2[NLAYER], *be2[NLAYER];
  const float *Wm[NLAYER], *bm[NLAYER], *Wh1[NLAYER], *bh1[NLAYER];
  const float *Wh2[NLAYER], *bh2[NLAYER], *Wx1[NLAYER], *bx1[NLAYER], *Wx2[NLAYER];
  int base = 7;
  for (int l = 0; l < NLAYER; ++l) {
    We1[l] = (const float*)d_in[base + 0];  be1[l] = (const float*)d_in[base + 1];
    We2[l] = (const float*)d_in[base + 2];  be2[l] = (const float*)d_in[base + 3];
    Wm [l] = (const float*)d_in[base + 4];  bm [l] = (const float*)d_in[base + 5];
    Wh1[l] = (const float*)d_in[base + 6];  bh1[l] = (const float*)d_in[base + 7];
    Wh2[l] = (const float*)d_in[base + 8];  bh2[l] = (const float*)d_in[base + 9];
    Wx1[l] = (const float*)d_in[base + 10]; bx1[l] = (const float*)d_in[base + 11];
    Wx2[l] = (const float*)d_in[base + 12];
    base += 13;
  }
  const float* decW1 = (const float*)d_in[base + 0];
  const float* decb1 = (const float*)d_in[base + 1];
  const float* decW2 = (const float*)d_in[base + 2];
  const float* decb2 = (const float*)d_in[base + 3];

  // workspace layout
  char*  ws   = (char*)d_ws;
  size_t off  = 0;
  float* hbuf = (float*)(ws + off); off += (size_t)NNODE * HID * 4;
  float* xbuf = (float*)(ws + off); off += (size_t)NNODE * 4 * 4;
  float* wmb  = (float*)(ws + off); off += (size_t)NNODE * HID * 4;
  float* numb = (float*)(ws + off); off += (size_t)NNODE * 4 * 4;
  _Float16* h16b = (_Float16*)(ws + off); off += (size_t)NNODE * HID * 2;
  _Float16* pk   = (_Float16*)(ws + off);
  const size_t SZ55 = (size_t)5 * 5 * 512;   // 160xK weights (We1, Wh1)
  const size_t SZ35 = (size_t)3 * 5 * 512;   // 96xK weights  (We2, Wh2, Wx1)
  _Float16 *We1p[NLAYER], *We2p[NLAYER], *Wh1p[NLAYER], *Wh2p[NLAYER], *Wx1p[NLAYER];
  {
    _Float16* p = pk;
    for (int l = 0; l < NLAYER; ++l) {
      We1p[l] = p; p += SZ55;
      We2p[l] = p; p += SZ35;
      Wh1p[l] = p; p += SZ55;
      Wh2p[l] = p; p += SZ35;
      Wx1p[l] = p; p += SZ35;
    }
  }

  // pack all weights to f16 fragment layout
  for (int l = 0; l < NLAYER; ++l) {
    pack_wmma_b<<<(int)((SZ55 + 255) / 256), 256, 0, stream>>>(We1[l], We1p[l], 146, HID, 5, 5);
    pack_wmma_b<<<(int)((SZ35 + 255) / 256), 256, 0, stream>>>(We2[l], We2p[l], HID, HID, 3, 5);
    pack_wmma_b<<<(int)((SZ55 + 255) / 256), 256, 0, stream>>>(Wh1[l], Wh1p[l], 2 * HID, HID, 5, 5);
    pack_wmma_b<<<(int)((SZ35 + 255) / 256), 256, 0, stream>>>(Wh2[l], Wh2p[l], HID, HID, 3, 5);
    pack_wmma_b<<<(int)((SZ35 + 255) / 256), 256, 0, stream>>>(Wx1[l], Wx1p[l], HID, HID, 3, 5);
  }

  embed_kernel<<<(NNODE * HID + 255) / 256, 256, 0, stream>>>(
      scalars, mom4, embW, embb, hbuf, h16b, xbuf);

  for (int l = 0; l < NLAYER; ++l) {
    int cx = (l < NLAYER - 1) ? 1 : 0;
    lgeb_edge_kernel<<<NNODE, 128, 0, stream>>>(
        h16b, xbuf, edge_j, We1p[l], We2p[l], Wx1p[l],
        be1[l], be2[l], Wm[l], bm[l], bx1[l], Wx2[l], wmb, numb, cx);
    lgeb_node_kernel<<<NNODE / 64, 128, 0, stream>>>(
        hbuf, h16b, xbuf, wmb, numb, Wh1p[l], Wh2p[l], bh1[l], bh2[l], cx);
  }

  decoder_kernel<<<1, 256, 0, stream>>>(hbuf, mask, decW1, decb1, decW2, decb2, (float*)d_out);
}
